// Attn_58995670777930
// MI455X (gfx1250) — matrix operand order repaired and checked by reading, not compiled, not run on hardware
//
#include <hip/hip_runtime.h>
#include <hip/hip_bf16.h>
#include <math.h>

typedef __attribute__((ext_vector_type(16))) _Float16 v16h;
typedef __attribute__((ext_vector_type(8)))  _Float16 v8h;
typedef __attribute__((ext_vector_type(8)))  float    v8f;

#define BATCH 4
#define CCH   256
#define NPIX  4096
#define DQK   32

union HalfVec { v16h v; v8h h[2]; };

// ---------------------------------------------------------------------------
// Kernel 1: q/k projections, written directly in WMMA fragment layouts.
// A-fragment (16x32 f16): lane l, half j -> m = l&15, k = (j&7) + 8*(l>>4) + 16*(j>>3)
// B-fragment (32x16 f16): lane l, half j -> n = l&15, k = (j&7) + 8*(l>>4) + 16*(j>>3)
// ---------------------------------------------------------------------------
__global__ __launch_bounds__(256) void pack_qk_kernel(
    const float* __restrict__ x,  const float* __restrict__ wq, const float* __restrict__ bq,
    const float* __restrict__ wk, const float* __restrict__ bk,
    _Float16* __restrict__ qpack, _Float16* __restrict__ kpack)
{
    __shared__ float xs[CCH * 16];
    const int b   = blockIdx.x >> 8;
    const int t   = blockIdx.x & 255;     // n-tile of 16
    const int tid = threadIdx.x;

    const float* xb = x + ((size_t)b * CCH) * NPIX + t * 16;
    for (int i = tid; i < CCH * 16; i += 256) {
        const int c = i >> 4, nn = i & 15;
        xs[i] = xb[(size_t)c * NPIX + nn];
    }
    __syncthreads();

    for (int s = tid; s < 512; s += 256) {
        const int l  = s >> 4, j = s & 15;
        const int nn = l & 15;
        const int hg = l >> 4;
        // Q -> A-fragment
        {
            const int d = (j & 7) + 8 * hg + 16 * (j >> 3);
            float acc = bq[d];
            const float* w = wq + d * CCH;
            for (int c = 0; c < CCH; ++c) acc += xs[c * 16 + nn] * w[c];
            qpack[(((size_t)b * 256 + t) * 32 + l) * 16 + j] = (_Float16)acc;
        }
        // K -> B-fragment
        {
            const int d = (j & 7) + 8 * hg + 16 * (j >> 3);
            float acc = bk[d];
            const float* w = wk + d * CCH;
            for (int c = 0; c < CCH; ++c) acc += xs[c * 16 + nn] * w[c];
            kpack[(((size_t)b * 256 + t) * 32 + l) * 16 + j] = (_Float16)acc;
        }
    }
}

// ---------------------------------------------------------------------------
// Kernel 2: v projection -> V^T in B-fragment layout for the AV gemm.
// vpack[((b*128+kt)*16+ct)*512 + l*16 + j] = v[b, c=ct*16+(l&15), n=kt*32+(j&7)+8*(l>>4)+16*(j>>3)]
// ---------------------------------------------------------------------------
__global__ __launch_bounds__(256) void pack_v_kernel(
    const float* __restrict__ x, const float* __restrict__ wv, const float* __restrict__ bv,
    _Float16* __restrict__ vpack)
{
    __shared__ float xs[CCH * 32];
    const int b   = blockIdx.x >> 7;
    const int kt  = blockIdx.x & 127;     // n-tile of 32
    const int tid = threadIdx.x;

    const float* xb = x + ((size_t)b * CCH) * NPIX + kt * 32;
    for (int i = tid; i < CCH * 32; i += 256) {
        const int c = i >> 5, nn = i & 31;
        xs[i] = xb[(size_t)c * NPIX + nn];
    }
    __syncthreads();

    for (int s = tid; s < 16 * 512; s += 256) {
        const int ct = s >> 9;
        const int r  = s & 511;
        const int l  = r >> 4, j = r & 15;
        const int c  = ct * 16 + (l & 15);
        const int nn = (j & 7) + 8 * (l >> 4) + 16 * (j >> 3);
        float acc = bv[c];
        const float* w = wv + c * CCH;
        for (int cc = 0; cc < CCH; ++cc) acc += xs[cc * 32 + nn] * w[cc];
        vpack[((((size_t)b * 128 + kt) * 16 + ct) * 512) + l * 16 + j] = (_Float16)acc;
    }
}

// ---------------------------------------------------------------------------
// Kernel 3 (FUSED): S = Q*K + row softmax, fully register-resident.
// Block = 512 threads = 16 waves, handles one (b, mt) strip of 16 query rows.
// Wave w owns columns [w*256, w*256+256) as 16 WMMA C-fragments (128 VGPRs).
// Row stats: shfl_xor(width=16) within wave halves (matches C layout rows
// r+8*hg), then a 1KB LDS array across the 16 waves. Final attn written once:
// f32 (output) + f16 (for AV gemm).
// ---------------------------------------------------------------------------
__global__ __launch_bounds__(512) void qk_softmax_fused_kernel(
    const _Float16* __restrict__ qpack, const _Float16* __restrict__ kpack,
    float* __restrict__ attn, _Float16* __restrict__ attn_h)
{
    __shared__ float partial[16][16];   // [wave][row]
    __shared__ float rowmaxs[16];
    __shared__ float rowinvs[16];

    const int tid  = threadIdx.x;
    const int w    = tid >> 5;          // wave 0..15
    const int lane = tid & 31;
    const int b    = blockIdx.x >> 8;
    const int mt   = blockIdx.x & 255;
    const int hg   = lane >> 4, nl = lane & 15;

    // A-fragment for this 16-row strip (shared by all column tiles)
    const v16h a = *(const v16h*)(qpack + (((size_t)b * 256 + mt) * 32 + lane) * 16);

    // 16 score tiles per wave: columns (w*16+i)*16 .. +15
    v8f cf[16];
    const _Float16* kbase = kpack + (((size_t)b * 256 + w * 16) * 32 + lane) * 16;
    #pragma unroll
    for (int i = 0; i < 16; ++i) {
        const v16h kf = *(const v16h*)(kbase + (size_t)i * (32 * 16));
        v8f z = {};
        cf[i] = __builtin_amdgcn_wmma_f32_16x16x32_f16(false, a, false, kf, (short)0, z, false, false);
    }

    // ---- row max ----
    {
        float lm[8];
        #pragma unroll
        for (int r = 0; r < 8; ++r) {
            float m = cf[0][r];
            #pragma unroll
            for (int i = 1; i < 16; ++i) m = fmaxf(m, cf[i][r]);
            #pragma unroll
            for (int off = 1; off < 16; off <<= 1)
                m = fmaxf(m, __shfl_xor(m, off, 16));
            lm[r] = m;
        }
        if (nl == 0) {
            #pragma unroll
            for (int r = 0; r < 8; ++r) partial[w][hg * 8 + r] = lm[r];
        }
    }
    __syncthreads();
    if (tid < 16) {
        float m = partial[0][tid];
        #pragma unroll
        for (int wv = 1; wv < 16; ++wv) m = fmaxf(m, partial[wv][tid]);
        rowmaxs[tid] = m;
    }
    __syncthreads();

    // ---- exp + row sum ----
    float rmax[8];
    #pragma unroll
    for (int r = 0; r < 8; ++r) rmax[r] = rowmaxs[hg * 8 + r];

    {
        float ls[8] = {0.f, 0.f, 0.f, 0.f, 0.f, 0.f, 0.f, 0.f};
        #pragma unroll
        for (int i = 0; i < 16; ++i) {
            #pragma unroll
            for (int r = 0; r < 8; ++r) {
                const float e = __expf(cf[i][r] - rmax[r]);
                cf[i][r] = e;
                ls[r] += e;
            }
        }
        #pragma unroll
        for (int r = 0; r < 8; ++r) {
            float s = ls[r];
            #pragma unroll
            for (int off = 1; off < 16; off <<= 1)
                s += __shfl_xor(s, off, 16);
            ls[r] = s;
        }
        __syncthreads();   // partial[] reuse: all prior readers are done
        if (nl == 0) {
            #pragma unroll
            for (int r = 0; r < 8; ++r) partial[w][hg * 8 + r] = ls[r];
        }
    }
    __syncthreads();
    if (tid < 16) {
        float s = partial[0][tid];
        #pragma unroll
        for (int wv = 1; wv < 16; ++wv) s += partial[wv][tid];
        rowinvs[tid] = 1.0f / s;
    }
    __syncthreads();

    // ---- normalize + store (only memory traffic of this kernel's output) ----
    float rinv[8];
    #pragma unroll
    for (int r = 0; r < 8; ++r) rinv[r] = rowinvs[hg * 8 + r];

    const size_t rowbase = ((size_t)b * NPIX + (size_t)mt * 16) * NPIX;
    #pragma unroll
    for (int i = 0; i < 16; ++i) {
        const int col = (w * 16 + i) * 16 + nl;
        #pragma unroll
        for (int r = 0; r < 8; ++r) {
            const int row = hg * 8 + r;
            const float v = cf[i][r] * rinv[r];
            const size_t idx = rowbase + (size_t)row * NPIX + col;
            attn[idx]   = v;
            attn_h[idx] = (_Float16)v;
        }
    }
}

// ---------------------------------------------------------------------------
// Kernel 4: Om[m,c] = sum_n attn[m,n] * v[c,n]; out[b,c,m] = g*Om + x[b,c,m].
// Each wave: 16(m) x 64(c) tile, 128 K-steps of 32, 4 WMMAs per step.
// attn_h re-reads across the 4 c-groups are L2-resident (134MB < 192MB L2).
// ---------------------------------------------------------------------------
__global__ __launch_bounds__(256) void av_gemm_kernel(
    const _Float16* __restrict__ attn_h, const _Float16* __restrict__ vpack,
    const float* __restrict__ x, const float* __restrict__ gamma,
    float* __restrict__ out)
{
    const int w    = blockIdx.x * 8 + (threadIdx.x >> 5);
    const int lane = threadIdx.x & 31;
    const int b  = w >> 10;
    const int mt = (w >> 2) & 255;
    const int cg = w & 3;                     // group of 4 c-tiles (64 channels)
    const int hg = lane >> 4, nl = lane & 15;

    v8f acc[4] = {};

    // A-fragment rows come straight from the f16 attn copy: two 16B loads/step.
    const _Float16* arow  = attn_h + ((size_t)b * NPIX + (size_t)(mt * 16 + nl)) * NPIX + 8 * hg;
    const _Float16* vbase = vpack + ((((size_t)b * 128) * 16) + cg * 4) * 512 + lane * 16;

    #pragma unroll 2
    for (int kt = 0; kt < 128; ++kt) {
        HalfVec A;
        A.h[0] = *(const v8h*)(arow + kt * 32);
        A.h[1] = *(const v8h*)(arow + kt * 32 + 16);
        const _Float16* vb = vbase + (size_t)kt * (16 * 512);
        #pragma unroll
        for (int ct = 0; ct < 4; ++ct) {
            const v16h Bf = *(const v16h*)(vb + ct * 512);
            acc[ct] = __builtin_amdgcn_wmma_f32_16x16x32_f16(
                false, A.v, false, Bf, (short)0, acc[ct], false, false);
        }
    }

    const float g = gamma[0];
    #pragma unroll
    for (int ct = 0; ct < 4; ++ct) {
        const int c = (cg * 4 + ct) * 16 + nl;
        const size_t base = ((size_t)b * CCH + c) * NPIX;
        #pragma unroll
        for (int r = 0; r < 8; ++r) {
            const int m = mt * 16 + r + 8 * hg;
            out[base + m] = g * acc[ct][r] + x[base + m];
        }
    }
}

// ---------------------------------------------------------------------------
extern "C" void kernel_launch(void* const* d_in, const int* in_sizes, int n_in,
                              void* d_out, int out_size, void* d_ws, size_t ws_size,
                              hipStream_t stream)
{
    (void)in_sizes; (void)n_in; (void)out_size; (void)ws_size;

    const float* x     = (const float*)d_in[0];
    const float* wq    = (const float*)d_in[1];
    const float* bq    = (const float*)d_in[2];
    const float* wk    = (const float*)d_in[3];
    const float* bk    = (const float*)d_in[4];
    const float* wv    = (const float*)d_in[5];
    const float* bv    = (const float*)d_in[6];
    const float* gamma = (const float*)d_in[7];

    float* out  = (float*)d_out;                              // [B,C,W,H]
    float* attn = out + (size_t)BATCH * CCH * NPIX;           // [B,N,N] output #2

    // workspace: attn_h (134MB) | qpack (1MB) | kpack (1MB) | vpack (8MB)
    _Float16* attn_h = (_Float16*)d_ws;
    _Float16* qpack  = attn_h + (size_t)BATCH * NPIX * NPIX;
    _Float16* kpack  = qpack  + (size_t)BATCH * NPIX * DQK;
    _Float16* vpack  = kpack  + (size_t)BATCH * NPIX * DQK;

    pack_qk_kernel<<<BATCH * 256, 256, 0, stream>>>(x, wq, bq, wk, bk, qpack, kpack);
    pack_v_kernel <<<BATCH * 128, 256, 0, stream>>>(x, wv, bv, vpack);
    qk_softmax_fused_kernel<<<BATCH * 256, 512, 0, stream>>>(qpack, kpack, attn, attn_h);
    av_gemm_kernel<<<(BATCH * 256 * 4) / 8, 256, 0, stream>>>(attn_h, vpack, x, gamma, out);
}
